// snn_test_17051020165233
// MI455X (gfx1250) — compile-verified, matrix operand order checked by reading, not run on hardware
//
#include <hip/hip_runtime.h>
#include <math.h>

// ---------------------------------------------------------------------------
// Problem constants (from reference)
// ---------------------------------------------------------------------------
#define BB 64
#define TT 256
#define FF 1024
#define HH 512
#define TOPK 17          // T//16 + 1
#define BETA 0.9f
#define THR  1.0f

typedef __bf16 bf16_t;
typedef __attribute__((ext_vector_type(16))) __bf16 v16bf;
typedef __attribute__((ext_vector_type(8)))  __bf16 v8bf;
typedef __attribute__((ext_vector_type(8)))  float   v8f;

// ---------------------------------------------------------------------------
// f32 -> bf16 elementwise convert
// ---------------------------------------------------------------------------
__global__ void cvt_f32_bf16(const float* __restrict__ in, bf16_t* __restrict__ out, int n) {
    int i = blockIdx.x * blockDim.x + threadIdx.x;
    if (i < n) out[i] = (bf16_t)in[i];
}

// ---------------------------------------------------------------------------
// Fold Wc = W1 @ Wf  (512x1024, bf16 out), bc = W1 @ bf + b1 (f32)
// ---------------------------------------------------------------------------
__global__ void fuse_weights(const float* __restrict__ W1, const float* __restrict__ Wf,
                             const float* __restrict__ bfv, const float* __restrict__ b1,
                             bf16_t* __restrict__ Wc, float* __restrict__ bc) {
    int f = blockIdx.x * blockDim.x + threadIdx.x;   // 0..FF-1
    int h = blockIdx.y;                              // 0..HH-1
    float acc = 0.f;
    for (int j = 0; j < HH; ++j) acc += W1[h * HH + j] * Wf[(size_t)j * FF + f];
    Wc[(size_t)h * FF + f] = (bf16_t)acc;
    if (f == 0) {
        float a = b1[h];
        for (int j = 0; j < HH; ++j) a += W1[h * HH + j] * bfv[j];
        bc[h] = a;
    }
}

// ---------------------------------------------------------------------------
// WMMA bf16 fragment loader.
// 16-bit A/B operand layout (CDNA5 ISA §7.12.2):
//   lanes 0-15  hold row r, K = {0..7} in V0..V3 and {16..23} in V4..V7
//   lanes 16-31 hold row r, K = {8..15}           and {24..31}
// -> per lane: two contiguous 16B (b128) chunks at halfword offsets sel*8 and sel*8+16.
// ---------------------------------------------------------------------------
__device__ inline v16bf load_frag(const bf16_t* __restrict__ base, int ld,
                                  int row0, int k0, int r, int sel) {
    const bf16_t* p = base + (size_t)(row0 + r) * ld + k0 + sel * 8;
    v8bf lo = *(const v8bf*)(p);
    v8bf hi = *(const v8bf*)(p + 16);
    return __builtin_shufflevector(lo, hi, 0,1,2,3,4,5,6,7,8,9,10,11,12,13,14,15);
}

// ---------------------------------------------------------------------------
// C = A[M,K](bf16,row-major) @ Bw[N,K]^T(bf16,row-major) + bias[N]
// Block: 256 threads = 8 waves (4 M x 2 N). Wave tile 64x64 = 4x4 WMMA tiles.
// WG tile: 256(M) x 128(N). Requires M%256==0, N%128==0, K%32==0.
// Prefetch is UNconditional: speculative prefetch past the tensor end never
// faults (ISA §10.5) and keeping it branch-free avoids the compiler cloning
// the hot loop into a branchy variant.
// ---------------------------------------------------------------------------
template<bool RELU, bool OUT_BF16>
__global__ __launch_bounds__(256)
void gemm_bf16_wmma(const bf16_t* __restrict__ A, const bf16_t* __restrict__ Bw,
                    const float* __restrict__ bias,
                    float* __restrict__ Cf, bf16_t* __restrict__ Cb,
                    int M, int N, int K) {
    const int lane = threadIdx.x & 31;
    const int wave = threadIdx.x >> 5;
    const int wm   = wave >> 1;            // 0..3
    const int wn   = wave & 1;             // 0..1
    const int r    = lane & 15;
    const int sel  = lane >> 4;
    const int row0 = blockIdx.y * 256 + wm * 64;
    const int col0 = blockIdx.x * 128 + wn * 64;

    v8f acc[4][4] = {};

    for (int k0 = 0; k0 < K; k0 += 32) {
        v16bf a[4], b[4];
#pragma unroll
        for (int i = 0; i < 4; ++i) {
            a[i] = load_frag(A, K, row0 + i * 16, k0, r, sel);
            // gfx1250 global_prefetch_b8 of next K-slab (speculative, branch-free)
            __builtin_prefetch(A + (size_t)(row0 + i * 16 + r) * K + k0 + 32, 0, 1);
        }
#pragma unroll
        for (int j = 0; j < 4; ++j)
            b[j] = load_frag(Bw, K, col0 + j * 16, k0, r, sel);
#pragma unroll
        for (int i = 0; i < 4; ++i)
#pragma unroll
            for (int j = 0; j < 4; ++j)
                acc[i][j] = __builtin_amdgcn_wmma_f32_16x16x32_bf16(
                    false, a[i], false, b[j], (short)0, acc[i][j], false, false);
    }

#pragma unroll
    for (int i = 0; i < 4; ++i) {
#pragma unroll
        for (int j = 0; j < 4; ++j) {
            const int cc = col0 + j * 16 + r;
            const float bval = bias[cc];
#pragma unroll
            for (int e = 0; e < 8; ++e) {
                const int rr = row0 + i * 16 + e + sel * 8;   // C/D layout: V[e] -> M=e / M=8+e
                float v = acc[i][j][e] + bval;
                if (RELU) v = v > 0.f ? v : 0.f;
                if (OUT_BF16) Cb[(size_t)rr * N + cc] = (bf16_t)v;
                else          Cf[(size_t)rr * N + cc] = v;
            }
        }
    }
}

// ---------------------------------------------------------------------------
// LIF scan: per (b,h) lane, sequential over T.
// m = BETA*m + x_t - s_prev*THR ; s = (m > THR) ; emit s (bf16 exact 0/1).
// (reset spike of the reference equals previous output spike)
// ---------------------------------------------------------------------------
__global__ void scan_lif(const float* __restrict__ X, bf16_t* __restrict__ S) {
    int id = blockIdx.x * blockDim.x + threadIdx.x;   // 0..BB*HH-1
    int b = id / HH, h = id % HH;
    const float* xp = X + (size_t)b * TT * HH + h;
    bf16_t*      sp = S + (size_t)b * TT * HH + h;
    float m = 0.f, sprev = 0.f;
    for (int t = 0; t < TT; ++t) {
        float x = xp[(size_t)t * HH];
        m = BETA * m + x - sprev * THR;
        float s = (m > THR) ? 1.0f : 0.0f;
        sp[(size_t)t * HH] = (bf16_t)s;
        sprev = s;
    }
}

// ---------------------------------------------------------------------------
// MIL head: logits[row] = sigmoid( (relu-out Z1[row,:] @ Mw2.T + Mb2) @ Mw3 + Mb3 )
// Mw2 (32x512 bf16) staged in LDS; lane j owns output channel j; wave32 reduce.
// ---------------------------------------------------------------------------
__global__ __launch_bounds__(256)
void head_kernel(const bf16_t* __restrict__ Z1, const bf16_t* __restrict__ Mw2b,
                 const float* __restrict__ Mb2, const float* __restrict__ Mw3,
                 const float* __restrict__ Mb3, float* __restrict__ logits) {
    __shared__ bf16_t sW[32 * HH];
    for (int i = threadIdx.x; i < 32 * HH; i += blockDim.x) sW[i] = Mw2b[i];
    __syncthreads();

    const int wave = threadIdx.x >> 5;
    const int lane = threadIdx.x & 31;
    const int row  = blockIdx.x * 8 + wave;           // 8 rows per block
    const bf16_t* zp = Z1 + (size_t)row * HH;
    const bf16_t* wp = sW + lane * HH;

    float acc = 0.f;
    for (int k = 0; k < HH; k += 8) {
#pragma unroll
        for (int u = 0; u < 8; ++u)
            acc += (float)zp[k + u] * (float)wp[k + u];
    }
    float contrib = (acc + Mb2[lane]) * Mw3[lane];
#pragma unroll
    for (int off = 16; off >= 1; off >>= 1)
        contrib += __shfl_xor(contrib, off, 32);
    if (lane == 0)
        logits[row] = 1.0f / (1.0f + __expf(-(contrib + Mb3[0])));
}

// ---------------------------------------------------------------------------
// top-17 mean of logits[b, 0..T) per clip
// ---------------------------------------------------------------------------
__global__ void topk_mean(const float* __restrict__ logits, float* __restrict__ out) {
    __shared__ float vals[TT];
    __shared__ float rmax[TT];
    __shared__ int   ridx[TT];
    const int b = blockIdx.x, t = threadIdx.x;
    vals[t] = logits[b * TT + t];
    __syncthreads();
    float sum = 0.f;
    for (int it = 0; it < TOPK; ++it) {
        rmax[t] = vals[t]; ridx[t] = t;
        __syncthreads();
        for (int s2 = TT / 2; s2 > 0; s2 >>= 1) {
            if (t < s2 && rmax[t + s2] > rmax[t]) { rmax[t] = rmax[t + s2]; ridx[t] = ridx[t + s2]; }
            __syncthreads();
        }
        if (t == 0) { sum += rmax[0]; vals[ridx[0]] = -1e30f; }
        __syncthreads();
    }
    if (t == 0) out[b] = sum / (float)TOPK;
}

// ---------------------------------------------------------------------------
// Host orchestration
// ---------------------------------------------------------------------------
extern "C" void kernel_launch(void* const* d_in, const int* in_sizes, int n_in,
                              void* d_out, int out_size, void* d_ws, size_t ws_size,
                              hipStream_t stream) {
    (void)in_sizes; (void)n_in; (void)out_size; (void)ws_size;
    const float* f_f = (const float*)d_in[0];
    // d_in[1] = seq_len (int64) — always T, unused
    const float* Wf  = (const float*)d_in[2];
    const float* bfv = (const float*)d_in[3];
    const float* W1  = (const float*)d_in[4];
    const float* b1  = (const float*)d_in[5];
    const float* W2  = (const float*)d_in[6];
    const float* b2  = (const float*)d_in[7];
    const float* Mw1 = (const float*)d_in[8];
    const float* Mb1 = (const float*)d_in[9];
    const float* Mw2 = (const float*)d_in[10];
    const float* Mb2 = (const float*)d_in[11];
    const float* Mw3 = (const float*)d_in[12];
    const float* Mb3 = (const float*)d_in[13];
    float* out = (float*)d_out;

    // workspace carve-up (256B aligned); X2 aliases X1, Z1 aliases S1
    char* ws = (char*)d_ws;
    auto carve = [&](size_t bytes) { char* p = ws; ws += (bytes + 255) & ~(size_t)255; return p; };
    const size_t nBTF = (size_t)BB * TT * FF;     // 16.7M
    const size_t nBTH = (size_t)BB * TT * HH;     // 8.4M
    bf16_t* A_bf   = (bf16_t*)carve(nBTF * 2);          // f_f in bf16
    bf16_t* Wc_bf  = (bf16_t*)carve((size_t)HH * FF * 2);
    float*  bc     = (float*) carve((size_t)HH * 4);
    bf16_t* W2_bf  = (bf16_t*)carve((size_t)HH * HH * 2);
    bf16_t* Mw1_bf = (bf16_t*)carve((size_t)HH * HH * 2);
    bf16_t* Mw2_bf = (bf16_t*)carve((size_t)32 * HH * 2);
    float*  X1     = (float*) carve(nBTH * 4);          // reused for X2
    bf16_t* S1     = (bf16_t*)carve(nBTH * 2);          // reused for Z1
    bf16_t* SPK    = (bf16_t*)carve(nBTH * 2);
    float*  logits = (float*) carve((size_t)BB * TT * 4);
    float*  X2 = X1;
    bf16_t* Z1 = S1;

    // 1) precision converts
    cvt_f32_bf16<<<(unsigned)(nBTF / 256), 256, 0, stream>>>(f_f, A_bf, (int)nBTF);
    cvt_f32_bf16<<<(HH * HH) / 256, 256, 0, stream>>>(W2,  W2_bf,  HH * HH);
    cvt_f32_bf16<<<(HH * HH) / 256, 256, 0, stream>>>(Mw1, Mw1_bf, HH * HH);
    cvt_f32_bf16<<<(32 * HH) / 256, 256, 0, stream>>>(Mw2, Mw2_bf, 32 * HH);

    // 2) fold first two linears: Wc = W1@Wf, bc = W1@bf + b1
    fuse_weights<<<dim3(FF / 256, HH), 256, 0, stream>>>(W1, Wf, bfv, b1, Wc_bf, bc);

    // 3) X1 = f_f @ Wc^T + bc           (M=16384, N=512, K=1024)
    gemm_bf16_wmma<false, false><<<dim3(HH / 128, (BB * TT) / 256), 256, 0, stream>>>(
        A_bf, Wc_bf, bc, X1, nullptr, BB * TT, HH, FF);

    // 4) layer-1 LIF scan -> spikes s1
    scan_lif<<<(BB * HH) / 256, 256, 0, stream>>>(X1, S1);

    // 5) X2 = s1 @ W2^T + b2            (M=16384, N=512, K=512)
    gemm_bf16_wmma<false, false><<<dim3(HH / 128, (BB * TT) / 256), 256, 0, stream>>>(
        S1, W2_bf, b2, X2, nullptr, BB * TT, HH, HH);

    // 6) layer-2 LIF scan -> spikes spk
    scan_lif<<<(BB * HH) / 256, 256, 0, stream>>>(X2, SPK);

    // 7) Z1 = relu(spk @ Mw1^T + Mb1)   (bf16 out for the head)
    gemm_bf16_wmma<true, true><<<dim3(HH / 128, (BB * TT) / 256), 256, 0, stream>>>(
        SPK, Mw1_bf, Mb1, nullptr, Z1, BB * TT, HH, HH);

    // 8) head: 32-wide projection + sigmoid
    head_kernel<<<(BB * TT) / 8, 256, 0, stream>>>(Z1, Mw2_bf, Mb2, Mw3, Mb3, logits);

    // 9) top-17 mean per clip
    topk_mean<<<BB, TT, 0, stream>>>(logits, out);
}
